// MultiMemoryAttentionComparable_4647154614974
// MI455X (gfx1250) — matrix operand order repaired and checked by reading, not compiled, not run on hardware
//
#include <hip/hip_runtime.h>

// ---------------------------------------------------------------------------
// MI455X (gfx1250, wave32) implementation of multi-memory linear attention.
// All heavy math runs through v_wmma_f32_16x16x32_f16 (fp32 accumulate).
// Projection GEMMs are double-buffered: global_load_b128 of tile k+1 issues
// before the WMMAs of tile k, one s_barrier per K step.
// ---------------------------------------------------------------------------

typedef __attribute__((ext_vector_type(16))) _Float16 v16h;
typedef __attribute__((ext_vector_type(8)))  _Float16 v8h;
typedef __attribute__((ext_vector_type(8)))  float    v8f;

#define B_    2
#define S_    4096
#define D_    2048
#define H_    16
#define HD_   128
#define M_    4
#define NCHUNK 32
#define EPS_  1e-6f
#define SCALE_ 0.08838834764831845f   // 128^-0.5

__device__ __forceinline__ float elu1(float x) {
  return x > 0.f ? x + 1.f : __expf(x);   // elu(x)+1
}

// ---- WMMA fragment loaders from LDS (f16, padded row stride `ld` halves) ----
// A matrix 16x32 (MxK): lanes 0-15 hold K=0..7 & 16..23, lanes 16-31 hold
// K=8..15 & 24..31 (CDNA5 ISA 7.12.2). Two ds_load_b128 per fragment.
__device__ __forceinline__ v16h load_fragA(const _Float16* base, int rowBase,
                                           int ld, int kofs) {
  const int lane = threadIdx.x & 31;
  const _Float16* p = base + (rowBase + (lane & 15)) * ld + kofs +
                      ((lane < 16) ? 0 : 8);
  v8h lo = *(const v8h*)p;
  v8h hi = *(const v8h*)(p + 16);
  v16h f;
#pragma unroll
  for (int i = 0; i < 8; ++i) { f[i] = lo[i]; f[i + 8] = hi[i]; }
  return f;
}

// B matrix 32x16 (KxN): lane = N column (lane&15); lanes 0-15 hold K=0..7 & 16..23,
// lanes 16-31 hold K=8..15 & 24..31 (same order as A).
__device__ __forceinline__ v16h load_fragB(const _Float16* base, int rowBase,
                                           int ld, int kofs) {
  const int lane = threadIdx.x & 31;
  const _Float16* p = base + (rowBase + (lane & 15)) * ld + kofs +
                      ((lane < 16) ? 0 : 8);
  v8h lo = *(const v8h*)p;
  v8h hi = *(const v8h*)(p + 16);
  v16h f;
#pragma unroll
  for (int i = 0; i < 8; ++i) { f[i] = lo[i]; f[i + 8] = hi[i]; }
  return f;
}

__device__ __forceinline__ v8f wmma32(v16h a, v16h b, v8f c) {
  return __builtin_amdgcn_wmma_f32_16x16x32_f16(false, a, false, b,
                                                (short)0, c, false, false);
}

// ===========================================================================
// GEMM:  Y = X @ W^T   (X:[Mr x 2048] fp32, W:[2048 x 2048] fp32)
// mode 0: Y row-major [Mr x 2048]
// mode 1: Y scattered to [b, h, s, hd]  (Mr = B*S, N = H*HD)
// grid: (N/128, M/128, nW); block 256 (8 waves). blockIdx.z picks W/Y.
// Double-buffered LDS: one barrier per 32-wide K step; global loads of the
// next tile are issued before the WMMAs of the current tile.
// ===========================================================================
#define GLD 40   // 32 K halves + 8 pad -> 80B rows (16B aligned)
#define NKT (D_ / 32)

__global__ __launch_bounds__(256) void gemm_xwt_kernel(
    const float* __restrict__ X,
    const float* __restrict__ W0, const float* __restrict__ W1,
    const float* __restrict__ W2,
    float* __restrict__ Y0, float* __restrict__ Y1, float* __restrict__ Y2,
    int mode) {
  __shared__ _Float16 sA[2][128 * GLD];
  __shared__ _Float16 sB[2][128 * GLD];

  const int t    = threadIdx.x;
  const int wave = t >> 5;
  const int lane = t & 31;
  const int bn   = blockIdx.x;
  const int bm   = blockIdx.y;
  const float* W = (blockIdx.z == 0) ? W0 : (blockIdx.z == 1 ? W1 : W2);
  float*       Y = (blockIdx.z == 0) ? Y0 : (blockIdx.z == 1 ? Y1 : Y2);

  v8f acc[8];
#pragma unroll
  for (int i = 0; i < 8; ++i)
#pragma unroll
    for (int e = 0; e < 8; ++e) acc[i][e] = 0.f;

  const int lrow = t >> 3;       // 0..31
  const int lcol = (t & 7) * 4;  // 0..28
  const float* xRow = X + (size_t)(bm * 128 + lrow) * D_ + lcol;
  const float* wRow = W + (size_t)(bn * 128 + lrow) * D_ + lcol;

  // prologue: stage K tile 0 into buffer 0
  {
#pragma unroll
    for (int p = 0; p < 4; ++p) {
      const float4 xv = *(const float4*)(xRow + (size_t)p * 32 * D_);
      const float4 wv = *(const float4*)(wRow + (size_t)p * 32 * D_);
      _Float16* da = sA[0] + (lrow + p * 32) * GLD + lcol;
      da[0] = (_Float16)xv.x; da[1] = (_Float16)xv.y;
      da[2] = (_Float16)xv.z; da[3] = (_Float16)xv.w;
      _Float16* db = sB[0] + (lrow + p * 32) * GLD + lcol;
      db[0] = (_Float16)wv.x; db[1] = (_Float16)wv.y;
      db[2] = (_Float16)wv.z; db[3] = (_Float16)wv.w;
    }
  }

  for (int kt = 0; kt < NKT; ++kt) {
    const int cur = kt & 1;
    const int nxt = cur ^ 1;

    // issue global loads of the next tile before computing on the current one
    float4 xr[4], wr[4];
    const bool have_next = (kt + 1) < NKT;
    if (have_next) {
      const int k0n = (kt + 1) * 32;
#pragma unroll
      for (int p = 0; p < 4; ++p) {
        xr[p] = *(const float4*)(xRow + (size_t)p * 32 * D_ + k0n);
        wr[p] = *(const float4*)(wRow + (size_t)p * 32 * D_ + k0n);
      }
      if (kt + 2 < NKT) {  // global_prefetch_b8 of the tile after that
        __builtin_prefetch(xRow + (size_t)lrow * 0 + (kt + 2) * 32, 0, 1);
        __builtin_prefetch(wRow + (kt + 2) * 32, 0, 1);
      }
    }

    __syncthreads();   // stores of tile kt (prologue / previous iter) visible

    v16h a = load_fragA(sA[cur], wave * 16, GLD, 0);
#pragma unroll
    for (int nt = 0; nt < 8; ++nt) {
      v16h b = load_fragB(sB[cur], nt * 16, GLD, 0);
      acc[nt] = wmma32(a, b, acc[nt]);
    }

    if (have_next) {  // convert + stage next tile into the other buffer
#pragma unroll
      for (int p = 0; p < 4; ++p) {
        _Float16* da = sA[nxt] + (lrow + p * 32) * GLD + lcol;
        da[0] = (_Float16)xr[p].x; da[1] = (_Float16)xr[p].y;
        da[2] = (_Float16)xr[p].z; da[3] = (_Float16)xr[p].w;
        _Float16* db = sB[nxt] + (lrow + p * 32) * GLD + lcol;
        db[0] = (_Float16)wr[p].x; db[1] = (_Float16)wr[p].y;
        db[2] = (_Float16)wr[p].z; db[3] = (_Float16)wr[p].w;
      }
    }
  }

  // C/D layout: VGPR r -> M = r (lanes 0-15) or r+8 (lanes 16-31), N = lane&15
  const int rAdd = (lane < 16) ? 0 : 8;
  const int nFr  = lane & 15;
#pragma unroll
  for (int nt = 0; nt < 8; ++nt) {
#pragma unroll
    for (int r = 0; r < 8; ++r) {
      const int i = bm * 128 + wave * 16 + r + rAdd;
      const int n = bn * 128 + nt * 16 + nFr;
      const float val = acc[nt][r];
      if (mode == 0) {
        Y[(size_t)i * D_ + n] = val;
      } else {
        const int b = i >> 12, s = i & 4095;
        const int h = n >> 7,  hd = n & 127;
        Y[(((size_t)(b * H_ + h)) * S_ + s) * HD_ + hd] = val;
      }
    }
  }
}

// ===========================================================================
// qbar[b,h,d] = mean_s q[b,h,s,d]   (rel is linear in s -> mean commutes)
// ===========================================================================
__global__ __launch_bounds__(128) void qbar_kernel(const float* __restrict__ q,
                                                   float* __restrict__ qbar) {
  const int bh = blockIdx.x;     // B*H = 32 blocks
  const int d  = threadIdx.x;    // 128
  const float* p = q + (size_t)bh * S_ * HD_ + d;
  float s = 0.f;
  for (int ss = 0; ss < S_; ++ss) s += p[(size_t)ss * HD_];
  qbar[bh * HD_ + d] = s * (1.f / (float)S_);
}

// ===========================================================================
// Memory softmax weights: wout[m, b*H+h]
// ===========================================================================
__global__ __launch_bounds__(64) void relw_kernel(const float* __restrict__ qbar,
                                                  const float* __restrict__ mnorm,
                                                  float* __restrict__ wout) {
  __shared__ float semp[M_];
  const int t = threadIdx.x;
  if (t < M_) {
    float s = 0.f;
    for (int i = 0; i < H_ * HD_; ++i) s += mnorm[t * H_ * HD_ + i];
    semp[t] = s;
  }
  __syncthreads();
  if (t < B_ * H_) {
    const int h = t & (H_ - 1);
    float rel[M_]; bool emp[M_]; bool alle = true;
#pragma unroll
    for (int m = 0; m < M_; ++m) {
      emp[m] = semp[m] < 1e-6f;
      alle   = alle && emp[m];
      float s = 0.f;
      for (int d = 0; d < HD_; ++d)
        s += qbar[t * HD_ + d] * mnorm[(m * H_ + h) * HD_ + d];
      rel[m] = s * SCALE_;
    }
    float mx = -1e30f;
#pragma unroll
    for (int m = 0; m < M_; ++m) if (!emp[m] && rel[m] > mx) mx = rel[m];
    float w[M_]; float ssum = 0.f;
#pragma unroll
    for (int m = 0; m < M_; ++m) {
      w[m] = emp[m] ? 0.f : __expf(rel[m] - mx);
      ssum += w[m];
    }
#pragma unroll
    for (int m = 0; m < M_; ++m)
      wout[m * (B_ * H_) + t] = alle ? 0.f : (w[m] / ssum);
  }
}

// ===========================================================================
// Chunked causal linear attention + multi-memory retrieval + gated combine.
// One workgroup (256 thr = 8 waves) per (b,h); sequential scan over 32 chunks.
// S^T state lives in persistent WMMA accumulators; f16 mirror in LDS feeds
// the next chunk's qp @ S_prev GEMM. ~246 KB dynamic LDS (CDNA5: 320KB/WGP).
// ===========================================================================
#define LDH 136   // 128 + 8 pad halves -> 272B rows (16B aligned)

__global__ __launch_bounds__(256) void attn_kernel(
    const float* __restrict__ q, const float* __restrict__ k,
    const float* __restrict__ v, const float* __restrict__ memories,
    const float* __restrict__ mnorm, const float* __restrict__ wts,
    const float* __restrict__ gate, float* __restrict__ out) {
  extern __shared__ char smem[];
  _Float16* sQ  = (_Float16*)smem;         // qp   [i][d]  (A)
  _Float16* sK  = sQ  + 128 * LDH;         // kp   [j][d]  (B for scores)
  _Float16* sKT = sK  + 128 * LDH;         // kp^T [d][j]  (B for S-update)
  _Float16* sVT = sKT + 128 * LDH;         // v^T  [e][j]  (A for S-upd, B intra)
  _Float16* sST = sVT + 128 * LDH;         // S^T  [e][d]  (B for inter)
  _Float16* sSC = sST + 128 * LDH;         // masked scores [i][j] (A for intra)
  _Float16* sMT = sSC + 128 * LDH;         // mem^T [e][d] (B for retrieval)
  float* fpool = (float*)(sMT + 128 * LDH);
  float* Zst   = fpool;         // 128: running Z (exclusive prefix)
  float* zadd  = fpool + 128;   // 128: this chunk's column sums of kp
  float* denr  = fpool + 256;   // 128: 1/clip(den)
  float* normr = fpool + 384;   // 128: 1/clip(norm_m) (per-m)
  float* sMN   = fpool + 512;   // 128: memory_norms[m,h,:]

  const int bh = blockIdx.x;
  const int b  = bh >> 4, h = bh & 15;
  const int t  = threadIdx.x;
  const int wave = t >> 5, lane = t & 31;
  const int rAdd = (lane < 16) ? 0 : 8;
  const int nFr  = lane & 15;

  const float g = 1.f / (1.f + __expf(-gate[h]));
  float wmem[M_];
#pragma unroll
  for (int m = 0; m < M_; ++m) wmem[m] = wts[m * (B_ * H_) + bh];

  const float* qb = q + (size_t)bh * S_ * HD_;
  const float* kb = k + (size_t)bh * S_ * HD_;
  const float* vb = v + (size_t)bh * S_ * HD_;
  float* outp = out + (size_t)b * S_ * D_ + h * HD_;   // [b, s, h*128 + e]

  // init state
  for (int i = t; i < 128 * LDH; i += 256) sST[i] = (_Float16)0.f;
  if (t < 128) Zst[t] = 0.f;
  v8f accS[8];
#pragma unroll
  for (int i = 0; i < 8; ++i)
#pragma unroll
    for (int e = 0; e < 8; ++e) accS[i][e] = 0.f;
  __syncthreads();

  const int lrow = t >> 1;          // 0..127
  const int lcb  = (t & 1) * 64;    // 0 or 64

  for (int n = 0; n < NCHUNK; ++n) {
    const float* qc = qb + (size_t)n * 128 * HD_;
    const float* kc = kb + (size_t)n * 128 * HD_;
    const float* vc = vb + (size_t)n * 128 * HD_;

    // ---- P1: load chunk, apply elu+1, stage (incl. transposes) -------------
#pragma unroll
    for (int c4 = 0; c4 < 64; c4 += 4) {
      const int d0 = lcb + c4;
      const float4 qv = *(const float4*)(qc + (size_t)lrow * HD_ + d0);
      const float4 kv = *(const float4*)(kc + (size_t)lrow * HD_ + d0);
      const float4 vv = *(const float4*)(vc + (size_t)lrow * HD_ + d0);
      const float qp0 = elu1(qv.x), qp1 = elu1(qv.y), qp2 = elu1(qv.z), qp3 = elu1(qv.w);
      const float kp0 = elu1(kv.x), kp1 = elu1(kv.y), kp2 = elu1(kv.z), kp3 = elu1(kv.w);
      _Float16* dq = sQ + lrow * LDH + d0;
      dq[0] = (_Float16)qp0; dq[1] = (_Float16)qp1;
      dq[2] = (_Float16)qp2; dq[3] = (_Float16)qp3;
      _Float16* dk = sK + lrow * LDH + d0;
      dk[0] = (_Float16)kp0; dk[1] = (_Float16)kp1;
      dk[2] = (_Float16)kp2; dk[3] = (_Float16)kp3;
      sKT[(d0 + 0) * LDH + lrow] = (_Float16)kp0;
      sKT[(d0 + 1) * LDH + lrow] = (_Float16)kp1;
      sKT[(d0 + 2) * LDH + lrow] = (_Float16)kp2;
      sKT[(d0 + 3) * LDH + lrow] = (_Float16)kp3;
      sVT[(d0 + 0) * LDH + lrow] = (_Float16)vv.x;
      sVT[(d0 + 1) * LDH + lrow] = (_Float16)vv.y;
      sVT[(d0 + 2) * LDH + lrow] = (_Float16)vv.z;
      sVT[(d0 + 3) * LDH + lrow] = (_Float16)vv.w;
    }
    __syncthreads();

    // ---- P2: scores = qp @ kp^T, causal mask, stage to LDS -----------------
    v8f accT[8];
#pragma unroll
    for (int i = 0; i < 8; ++i)
#pragma unroll
      for (int e = 0; e < 8; ++e) accT[i][e] = 0.f;
#pragma unroll
    for (int ks = 0; ks < 4; ++ks) {
      v16h a = load_fragA(sQ, wave * 16, LDH, ks * 32);
#pragma unroll
      for (int nt = 0; nt < 8; ++nt) {
        v16h bf = load_fragB(sK, nt * 16, LDH, ks * 32);
        accT[nt] = wmma32(a, bf, accT[nt]);
      }
    }
#pragma unroll
    for (int nt = 0; nt < 8; ++nt)
#pragma unroll
      for (int r = 0; r < 8; ++r) {
        const int i = wave * 16 + r + rAdd;
        const int j = nt * 16 + nFr;
        const float val = (j <= i) ? accT[nt][r] : 0.f;
        sSC[i * LDH + j] = (_Float16)val;
      }
    __syncthreads();

    // ---- P3: denominators (Z_prev exclusive) and kp column sums ------------
    if (t < 128) {
      float id = 0.f, rs = 0.f;
      for (int d = 0; d < 128; ++d) id += (float)sQ[t * LDH + d] * Zst[d];
      for (int j = 0; j <= t; ++j)  rs += (float)sSC[t * LDH + j];
      float den = id + rs;
      denr[t] = 1.f / (den < EPS_ ? EPS_ : den);
    } else {
      const int d = t - 128;
      float s = 0.f;
      for (int j = 0; j < 128; ++j) s += (float)sKT[d * LDH + j];
      zadd[d] = s;
    }
    __syncthreads();

    // ---- P6: memory retrieval: sum_m w_m * (qp @ mem_m) / clip(qp.mnorm_m) -
    v8f macc[8];
#pragma unroll
    for (int i = 0; i < 8; ++i)
#pragma unroll
      for (int e = 0; e < 8; ++e) macc[i][e] = 0.f;
    for (int m = 0; m < M_; ++m) {
      const float* mb = memories + ((size_t)(m * H_ + h)) * HD_ * HD_;
#pragma unroll
      for (int c4 = 0; c4 < 64; c4 += 4) {     // stage mem^T (row d -> col d)
        const int e0 = lcb + c4;
        const float4 mv = *(const float4*)(mb + (size_t)lrow * HD_ + e0);
        sMT[(e0 + 0) * LDH + lrow] = (_Float16)mv.x;
        sMT[(e0 + 1) * LDH + lrow] = (_Float16)mv.y;
        sMT[(e0 + 2) * LDH + lrow] = (_Float16)mv.z;
        sMT[(e0 + 3) * LDH + lrow] = (_Float16)mv.w;
      }
      if (t < 128) sMN[t] = mnorm[(m * H_ + h) * HD_ + t];
      __syncthreads();
      if (t < 128) {
        float s = 0.f;
        for (int d = 0; d < 128; ++d) s += (float)sQ[t * LDH + d] * sMN[d];
        normr[t] = 1.f / (s < EPS_ ? EPS_ : s);
      }
      __syncthreads();
      v8f am[8];
#pragma unroll
      for (int i = 0; i < 8; ++i)
#pragma unroll
        for (int e = 0; e < 8; ++e) am[i][e] = 0.f;
#pragma unroll
      for (int ks = 0; ks < 4; ++ks) {
        v16h a = load_fragA(sQ, wave * 16, LDH, ks * 32);
#pragma unroll
        for (int nt = 0; nt < 8; ++nt) {
          v16h bf = load_fragB(sMT, nt * 16, LDH, ks * 32);
          am[nt] = wmma32(a, bf, am[nt]);
        }
      }
      const float wm = wmem[m];
#pragma unroll
      for (int nt = 0; nt < 8; ++nt)
#pragma unroll
        for (int r = 0; r < 8; ++r) {
          const int i = wave * 16 + r + rAdd;
          macc[nt][r] += wm * am[nt][r] * normr[i];
        }
      __syncthreads();   // before next m overwrites sMT/sMN/normr
    }

    // ---- P4: inter = qp @ S_prev ; += intra = scores @ v -------------------
#pragma unroll
    for (int i = 0; i < 8; ++i)
#pragma unroll
      for (int e = 0; e < 8; ++e) accT[i][e] = 0.f;
#pragma unroll
    for (int ks = 0; ks < 4; ++ks) {
      v16h a = load_fragA(sQ, wave * 16, LDH, ks * 32);
#pragma unroll
      for (int nt = 0; nt < 8; ++nt) {
        v16h bf = load_fragB(sST, nt * 16, LDH, ks * 32);
        accT[nt] = wmma32(a, bf, accT[nt]);
      }
    }
#pragma unroll
    for (int ks = 0; ks < 4; ++ks) {
      v16h a = load_fragA(sSC, wave * 16, LDH, ks * 32);
#pragma unroll
      for (int nt = 0; nt < 8; ++nt) {
        v16h bf = load_fragB(sVT, nt * 16, LDH, ks * 32);
        accT[nt] = wmma32(a, bf, accT[nt]);
      }
    }

    // ---- P7: gated combine, store to [b, s, h*128+e] ----------------------
#pragma unroll
    for (int nt = 0; nt < 8; ++nt)
#pragma unroll
      for (int r = 0; r < 8; ++r) {
        const int i = wave * 16 + r + rAdd;
        const int e = nt * 16 + nFr;
        const float local = accT[nt][r] * denr[i];
        outp[(size_t)(n * 128 + i) * D_ + e] = g * macc[nt][r] + (1.f - g) * local;
      }
    __syncthreads();   // all P4 reads of sST complete before update

    // ---- P5: S^T += v^T @ kp (persistent accumulators), mirror to LDS -----
#pragma unroll
    for (int ks = 0; ks < 4; ++ks) {
      v16h a = load_fragA(sVT, wave * 16, LDH, ks * 32);
#pragma unroll
      for (int nt = 0; nt < 8; ++nt) {
        v16h bf = load_fragB(sKT, nt * 16, LDH, ks * 32);
        accS[nt] = wmma32(a, bf, accS[nt]);
      }
    }
#pragma unroll
    for (int nt = 0; nt < 8; ++nt)
#pragma unroll
      for (int r = 0; r < 8; ++r) {
        const int e = wave * 16 + r + rAdd;
        const int d = nt * 16 + nFr;
        sST[e * LDH + d] = (_Float16)accS[nt][r];
      }
    if (t < 128) Zst[t] += zadd[t];
    __syncthreads();
  }
}

// ===========================================================================
extern "C" void kernel_launch(void* const* d_in, const int* in_sizes, int n_in,
                              void* d_out, int out_size, void* d_ws,
                              size_t ws_size, hipStream_t stream) {
  (void)in_sizes; (void)n_in; (void)out_size; (void)ws_size;
  const float* hidden = (const float*)d_in[0];
  const float* mems   = (const float*)d_in[1];
  const float* mnorm  = (const float*)d_in[2];
  const float* w_q    = (const float*)d_in[3];
  const float* w_k    = (const float*)d_in[4];
  const float* w_v    = (const float*)d_in[5];
  const float* w_o    = (const float*)d_in[6];
  const float* gate   = (const float*)d_in[7];
  float* y = (float*)d_out;

  const size_t nQKV = (size_t)B_ * H_ * S_ * HD_;   // 16.7M elements each
  float* qws  = (float*)d_ws;
  float* kws  = qws + nQKV;
  float* vws  = kws + nQKV;
  float* aout = vws + nQKV;                         // [B, S, D]
  float* qbar = aout + (size_t)B_ * S_ * D_;        // [B*H, HD]
  float* wts  = qbar + (size_t)B_ * H_ * HD_;       // [M, B*H]

  // 1) fused Q/K/V projections (grid.z selects weight/output)
  gemm_xwt_kernel<<<dim3(D_ / 128, (B_ * S_) / 128, 3), 256, 0, stream>>>(
      hidden, w_q, w_k, w_v, qws, kws, vws, /*mode=*/1);

  // 2) qbar = mean_s q   3) memory softmax weights
  qbar_kernel<<<B_ * H_, 128, 0, stream>>>(qws, qbar);
  relw_kernel<<<1, 64, 0, stream>>>(qbar, mnorm, wts);

  // 4) chunked causal linear attention + memory retrieval + gated combine
  const size_t dynLds = (size_t)7 * 128 * LDH * sizeof(_Float16)
                      + (size_t)5 * 128 * sizeof(float);   // ~246 KB
  attn_kernel<<<B_ * H_, 256, dynLds, stream>>>(qws, kws, vws, mems, mnorm,
                                                wts, gate, aout);

  // 5) output projection
  gemm_xwt_kernel<<<dim3(D_ / 128, (B_ * S_) / 128, 1), 256, 0, stream>>>(
      aout, w_o, w_o, w_o, y, y, y, /*mode=*/0);
}
